// BatchGraphAttentionLayer_32555852103748
// MI455X (gfx1250) — matrix operand order repaired and checked by reading, not compiled, not run on hardware
//
#include <hip/hip_runtime.h>
#include <math.h>

typedef __attribute__((ext_vector_type(16))) _Float16 v16h;
typedef __attribute__((ext_vector_type(8)))  _Float16 v8h;
typedef __attribute__((ext_vector_type(8)))  float    v8f;
typedef __attribute__((ext_vector_type(4)))  float    v4f;
typedef __attribute__((ext_vector_type(4)))  unsigned int v4u;

#define BATCH 16
#define NNODE 1024
#define FIN   128
#define FOUT  64
#define NEG_SLOPE 0.01f

// swizzled Wh (f16, WMMA-B-fragment order): [b][jc(32)][ct(4)][lane(32)][t(16)]
#define WSB_HALVES_PER_BATCH (32 * 4 * 32 * 16)   // 65536 halves = 128 KB
#define WSB_BYTES ((size_t)BATCH * WSB_HALVES_PER_BATCH * 2)  // 2 MB

// ---------------------------------------------------------------------------
// Kernel 1: Wh = h @ W (WMMA), store Wh pre-swizzled as f16 B-fragments,
//           compute f1 = Wh·a1, f2 = Wh·a2 in f32.
// grid = B * (N/128) = 128 blocks, 256 threads (8 waves), wave owns 16 rows.
// ---------------------------------------------------------------------------
__global__ __launch_bounds__(256) void gat_k1(
    const float* __restrict__ h, const float* __restrict__ W,
    const float* __restrict__ a,
    _Float16* __restrict__ wsB, float* __restrict__ f1w, float* __restrict__ f2w)
{
    __shared__ alignas(32) _Float16 ldsW[FIN * FOUT];  // 16 KB, fragment-ordered

    const int b    = blockIdx.x >> 3;
    const int i0   = (blockIdx.x & 7) * 128;
    const int tid  = threadIdx.x;
    const int wave = tid >> 5;
    const int lane = tid & 31;
    const int m16  = lane & 15;
    const int hi   = lane >> 4;

    // ---- stage W[b] (128x64 f32) into LDS as f16 in B-fragment order ----
    // element (k,n): chunk c=k/32, half hw=(k/8)%2, t=(k%8)+8*((k/16)%2); ct=n/16, ln=hw*16+n%16
    const float* Wb = W + (size_t)b * FIN * FOUT;
    #pragma unroll
    for (int s = 0; s < (FIN * FOUT) / 256; ++s) {
        int idx = tid + s * 256;
        int k = idx >> 6, n = idx & 63;
        int c = k >> 5, t = (k & 7) | (((k >> 4) & 1) << 3), hw = (k >> 3) & 1;
        int ct = n >> 4, ln = (hw << 4) | (n & 15);
        ldsW[(((((c << 2) + ct) << 5) | ln) << 4) | t] = (_Float16)Wb[idx];
    }
    __syncthreads();

    const int row = i0 + wave * 16 + m16;     // A-fragment row for this lane
    const float* hrow = h + ((size_t)b * NNODE + row) * FIN;

    v8f acc[4] = {};
    #pragma unroll
    for (int c = 0; c < 4; ++c) {             // K chunks of 32 over FIN=128
        // A fragment 16x32 f16: lane(m16,hi); elem t: K = c*32 + (t<8 ? hi*8+t : 16+hi*8+t-8)
        v4f h0 = *(const v4f*)(hrow + c * 32 + hi * 8);
        v4f h1 = *(const v4f*)(hrow + c * 32 + hi * 8 + 4);
        v4f h2 = *(const v4f*)(hrow + c * 32 + 16 + hi * 8);
        v4f h3 = *(const v4f*)(hrow + c * 32 + 16 + hi * 8 + 4);
        v16h af;
        #pragma unroll
        for (int t = 0; t < 4; ++t) {
            af[t]      = (_Float16)h0[t];
            af[t + 4]  = (_Float16)h1[t];
            af[t + 8]  = (_Float16)h2[t];
            af[t + 12] = (_Float16)h3[t];
        }
        #pragma unroll
        for (int ct = 0; ct < 4; ++ct) {      // 4 col tiles of 16 over FOUT=64
            v16h bf = *(const v16h*)&ldsW[((((c << 2) + ct) << 5) + lane) << 4];
            acc[ct] = __builtin_amdgcn_wmma_f32_16x16x32_f16(
                false, af, false, bf, (short)0, acc[ct], false, false);
        }
    }

    // ---- store Wh f16 pre-swizzled into B-fragment layout ----
    // C fragment: elem r of lane: M = hi*8 + r (within wave's 16 rows), N = m16
    // dest: jc = global 32-row chunk, hi selects lane half, t = hw*8+r (hw = wave&1)
    {
        const int jc = (i0 + wave * 16) >> 5;
        const int hw = wave & 1;
        const size_t fragBase = ((size_t)(b * 32 + jc)) * 4;
        #pragma unroll
        for (int ct = 0; ct < 4; ++ct) {
            v8h pk;
            #pragma unroll
            for (int r = 0; r < 8; ++r) pk[r] = (_Float16)acc[ct][r];
            size_t off = ((fragBase + ct) * 32 + (size_t)((hi << 4) | m16)) * 16 + hw * 8;
            *(v8h*)(wsB + off) = pk;          // 16B aligned contiguous store
        }
    }

    // ---- f1 = Wh·a1, f2 = Wh·a2 (f32, cross-lane reduction over 16 cols) ----
    {
        const float* ab = a + (size_t)b * 2 * FOUT;
        float a1v[4], a2v[4];
        #pragma unroll
        for (int ct = 0; ct < 4; ++ct) {
            a1v[ct] = ab[ct * 16 + m16];
            a2v[ct] = ab[FOUT + ct * 16 + m16];
        }
        #pragma unroll
        for (int r = 0; r < 8; ++r) {
            float p1 = 0.f, p2 = 0.f;
            #pragma unroll
            for (int ct = 0; ct < 4; ++ct) {
                float v = acc[ct][r];
                p1 += v * a1v[ct];
                p2 += v * a2v[ct];
            }
            #pragma unroll
            for (int off = 1; off < 16; off <<= 1) {   // reduce within each half-wave
                p1 += __shfl_xor(p1, off, 32);
                p2 += __shfl_xor(p2, off, 32);
            }
            if (m16 == 0) {
                int rr = i0 + wave * 16 + hi * 8 + r;
                f1w[b * NNODE + rr] = p1;
                f2w[b * NNODE + rr] = p2;
            }
        }
    }
}

// ---------------------------------------------------------------------------
// Kernel 2: fused softmax-attention: out = elu( softmax(lrelu(f1_i+f2_j)) @ Wh )
// Exact row max m_i = lrelu(f1_i + max_j f2_j) (monotonicity) -> single pass.
// The FULL per-batch Wh fragment buffer (128 KB) is staged into LDS once per
// block (CDNA5: 320 KB LDS/WGP), so the steady-state loop is LDS + exp + WMMA
// only; cuts L2 read traffic 8x vs per-wave global fragment reads.
// grid = B * (N/128) = 128 blocks, 256 threads, wave owns a 16x64 output tile.
// ---------------------------------------------------------------------------
__global__ __launch_bounds__(256) void gat_k2(
    const _Float16* __restrict__ wsB, const float* __restrict__ f1w,
    const float* __restrict__ f2w, float* __restrict__ out)
{
    __shared__ alignas(32) _Float16 ldsB[WSB_HALVES_PER_BATCH]; // 128 KB
    __shared__ alignas(16) float f2s[NNODE];                    // 4 KB
    __shared__ float red[8];

    const int b    = blockIdx.x >> 3;
    const int i0   = (blockIdx.x & 7) * 128;
    const int tid  = threadIdx.x;
    const int wave = tid >> 5;
    const int lane = tid & 31;
    const int m16  = lane & 15;
    const int hi   = lane >> 4;

    // ---- stage full Wh[b] fragment buffer into LDS (straight memcpy,
    //      coalesced global_load_b128 -> ds_store_b128) ----
    const _Float16* wb = wsB + (size_t)b * WSB_HALVES_PER_BATCH;
    {
        const v4u* src = (const v4u*)wb;
        v4u*       dst = (v4u*)ldsB;
        #pragma unroll
        for (int s = 0; s < (WSB_HALVES_PER_BATCH / 8) / 256; ++s)  // 32 x 16B/thread
            dst[tid + s * 256] = src[tid + s * 256];
    }

    // ---- stage f2[b] into LDS + block-wide max reduction ----
    float lmax = -1e30f;
    #pragma unroll
    for (int s = 0; s < NNODE / 256; ++s) {
        float v = f2w[b * NNODE + tid + s * 256];
        f2s[tid + s * 256] = v;
        lmax = fmaxf(lmax, v);
    }
    #pragma unroll
    for (int off = 1; off < 32; off <<= 1) lmax = fmaxf(lmax, __shfl_xor(lmax, off, 32));
    if (lane == 0) red[wave] = lmax;
    __syncthreads();                          // covers ldsB, f2s, red
    float mf2 = red[0];
    #pragma unroll
    for (int w = 1; w < 8; ++w) mf2 = fmaxf(mf2, red[w]);

    const int   row = i0 + wave * 16 + m16;     // lanes L and L+16 share a row
    const float f1r = f1w[b * NNODE + row];
    const float e0  = f1r + mf2;
    const float mrow = (e0 >= 0.f) ? e0 : NEG_SLOPE * e0;   // exact row max

    v8f acc[4] = {};
    float denom = 0.f;

    for (int jc = 0; jc < 32; ++jc) {           // K loop over all 1024 nodes
        const int j0 = jc * 32;
        // A-fragment p values: elem t -> j = j0 + (t<8 ? hi*8+t : 16+hi*8+(t-8))
        v4f q0 = *(const v4f*)&f2s[j0 + hi * 8];
        v4f q1 = *(const v4f*)&f2s[j0 + hi * 8 + 4];
        v4f q2 = *(const v4f*)&f2s[j0 + 16 + hi * 8];
        v4f q3 = *(const v4f*)&f2s[j0 + 16 + hi * 8 + 4];

        v16h af;
        #pragma unroll
        for (int t = 0; t < 16; ++t) {
            float f2v = (t < 4) ? q0[t] : (t < 8) ? q1[t - 4]
                      : (t < 12) ? q2[t - 8] : q3[t - 12];
            float e = f1r + f2v;
            e = (e >= 0.f) ? e : NEG_SLOPE * e;
            float p = __expf(e - mrow);         // p in (0,1]
            denom += p;
            af[t] = (_Float16)p;
        }

        const _Float16* fb = ldsB + (size_t)jc * 4 * 512;
        #pragma unroll
        for (int ct = 0; ct < 4; ++ct) {
            v16h bf = *(const v16h*)(fb + ct * 512 + lane * 16);  // ds_load_b128 x2
            acc[ct] = __builtin_amdgcn_wmma_f32_16x16x32_f16(
                false, af, false, bf, (short)0, acc[ct], false, false);
        }
    }

    // combine the two half-rows' denominators (lanes L and L+16 share a row)
    denom += __shfl_xor(denom, 16, 32);
    const float inv = 1.0f / denom;

    // epilogue: elu(acc/denom); C frag: elem r -> row i0+wave*16+hi*8+r, col ct*16+m16
    #pragma unroll
    for (int ct = 0; ct < 4; ++ct) {
        #pragma unroll
        for (int r = 0; r < 8; ++r) {
            float v = acc[ct][r] * inv;
            float o = (v > 0.f) ? v : expm1f(v);
            out[((size_t)b * NNODE + (i0 + wave * 16 + hi * 8 + r)) * FOUT
                + ct * 16 + m16] = o;
        }
    }
}

// ---------------------------------------------------------------------------
extern "C" void kernel_launch(void* const* d_in, const int* in_sizes, int n_in,
                              void* d_out, int out_size, void* d_ws, size_t ws_size,
                              hipStream_t stream)
{
    const float* h = (const float*)d_in[0];   // (16,1024,128) f32
    const float* W = (const float*)d_in[1];   // (16,128,64)  f32
    const float* a = (const float*)d_in[2];   // (16,1,128,1) f32
    float* outp = (float*)d_out;              // (16,1024,64) f32

    char* ws = (char*)d_ws;
    _Float16* wsB = (_Float16*)ws;                                     // 2 MB
    float* f1w = (float*)(ws + WSB_BYTES);                             // 64 KB
    float* f2w = (float*)(ws + WSB_BYTES + (size_t)BATCH * NNODE * 4); // 64 KB

    dim3 grid(BATCH * (NNODE / 128));
    dim3 block(256);
    gat_k1<<<grid, block, 0, stream>>>(h, W, a, wsB, f1w, f2w);
    gat_k2<<<grid, block, 0, stream>>>(wsB, f1w, f2w, outp);
}
